// RA_9818295239583
// MI455X (gfx1250) — matrix operand order repaired and checked by reading, not compiled, not run on hardware
//
#include <hip/hip_runtime.h>
#include <hip/hip_bf16.h>

#define HD    768
#define NHEAD 12
#define DHEAD 64
#define BSZ   8
#define TLEN  64
#define ILEN  197
#define PAIRS (TLEN * ILEN)        // 12608 per batch
#define MT    64                   // pairs per gate block (12608/64 = 197 exact)
#define GATE_TILES (PAIRS / MT)    // 197
#define KSTEP 32
#define NKSTEPS (HD / KSTEP)       // 24
#define XSTRIDE (HD + 8)           // 776 bf16 -> 1552B row stride, 2-sweep optimal
#define GATE_LDS_BYTES (MT * XSTRIDE * 2 + MT * 4)   // 99,328 + 256

typedef __bf16 bf16_t;
typedef __attribute__((ext_vector_type(8)))  __bf16 v8bf;
typedef __attribute__((ext_vector_type(16))) __bf16 v16bf;
typedef __attribute__((ext_vector_type(8)))  float  v8f;

union V16U { v16bf v; v8bf h[2]; };

// ---------------------------------------------------------------------------
// Kernel 0: W1 (fp32 [k][n]) -> W1t (bf16 [n][k]); B-fragments become
// two 16B chunks per lane, each contiguous along k.
// ---------------------------------------------------------------------------
__global__ __launch_bounds__(256) void k_transpose_w1(
    const float* __restrict__ W1, bf16_t* __restrict__ w1t) {
  int idx = blockIdx.x * 256 + threadIdx.x;     // 0 .. 768*768-1
  int n = idx / HD;
  int k = idx - n * HD;
  w1t[idx] = (bf16_t)W1[k * HD + n];            // coalesced write along k
}

// ---------------------------------------------------------------------------
// Kernel 1: generic fp32 projection  out[r][:] = in[r][:] @ W + bias
// 8 rows per block; input rows staged in LDS; W reads coalesced.
// ---------------------------------------------------------------------------
__global__ __launch_bounds__(256) void k_proj(
    const float* __restrict__ in, const float* __restrict__ W,
    const float* __restrict__ bias, float* __restrict__ out) {
  __shared__ float inS[8][HD];
  const int rb  = blockIdx.x * 8;
  const int tid = threadIdx.x;
  for (int idx = tid; idx < 8 * HD; idx += 256) {
    int rr = idx / HD;
    int k  = idx - rr * HD;
    inS[rr][k] = in[(size_t)(rb + rr) * HD + k];
  }
  __syncthreads();
  float acc[8][3];
#pragma unroll
  for (int rr = 0; rr < 8; ++rr) { acc[rr][0] = 0.f; acc[rr][1] = 0.f; acc[rr][2] = 0.f; }
  for (int k = 0; k < HD; ++k) {
    const float w0 = W[(size_t)k * HD + tid];
    const float w1 = W[(size_t)k * HD + tid + 256];
    const float w2 = W[(size_t)k * HD + tid + 512];
#pragma unroll
    for (int rr = 0; rr < 8; ++rr) {
      const float a = inS[rr][k];              // LDS broadcast
      acc[rr][0] += a * w0;
      acc[rr][1] += a * w1;
      acc[rr][2] += a * w2;
    }
  }
  const float b0 = bias[tid], b1v = bias[tid + 256], b2v = bias[tid + 512];
#pragma unroll
  for (int rr = 0; rr < 8; ++rr) {
    float* o = out + (size_t)(rb + rr) * HD;
    o[tid]       = acc[rr][0] + b0;
    o[tid + 256] = acc[rr][1] + b1v;
    o[tid + 512] = acc[rr][2] + b2v;
  }
}

// ---------------------------------------------------------------------------
// Kernel 2: fused gate MLP via bf16 WMMA.
// gate[b,t,i] = sigmoid( relu( (te[b,t]*ie[b,i]) @ W1 + b1 ) . W2 + b2 )
// Block: 512 threads (16 waves), 64 (t,i)-pairs (4 M-tiles), full 768 hidden.
// Each wave: 3 N-tiles x 4 M-tiles = 12 accumulators.
// ---------------------------------------------------------------------------
__global__ __launch_bounds__(512) void k_gate(
    const float* __restrict__ te, const float* __restrict__ ie,
    const bf16_t* __restrict__ w1t, const float* __restrict__ b1,
    const float* __restrict__ W2, const float* __restrict__ b2,
    float* __restrict__ gate) {
  extern __shared__ char smem_raw[];
  bf16_t* xs  = (bf16_t*)smem_raw;                          // [MT][XSTRIDE]
  float*  red = (float*)(smem_raw + MT * XSTRIDE * 2);      // [MT]

  const int b   = blockIdx.y;
  const int p0  = blockIdx.x * MT;
  const int tid = threadIdx.x;

  if (tid < MT) red[tid] = 0.f;

  // Build X = te[t] * ie[i] in fp32, store bf16 (row-major [m][k])
  for (int idx = tid; idx < MT * HD; idx += 512) {
    int m = idx / HD;
    int k = idx - m * HD;
    int p = p0 + m;
    int t = p / ILEN;
    int i = p - t * ILEN;
    float v = te[((size_t)(b * TLEN + t)) * HD + k] *
              ie[((size_t)(b * ILEN + i)) * HD + k];
    xs[m * XSTRIDE + k] = (bf16_t)v;
  }
  __syncthreads();

  const int wave  = tid >> 5;
  const int lane  = tid & 31;
  const int nbase = wave * 48;                 // 3 N-tiles of 16 per wave
  const int ncol  = lane & 15;
  // A fragment layout (ISA 16-bit A 16x32): lane half selects K{0-7,16-23}
  // vs K{8-15,24-31}, pairs of consecutive K per VGPR.
  const int mrow  = lane & 15;
  const int akofs = (lane >> 4) * 8;
  // B fragment layout (matches A): lane half selects K{0-7,16-23}
  // vs K{8-15,24-31}, pairs of consecutive K per VGPR.
  const int bkofs = (lane >> 4) * 8;

  v8f acc[4][3];
  const v8f vzero = {0.f, 0.f, 0.f, 0.f, 0.f, 0.f, 0.f, 0.f};
#pragma unroll
  for (int mt = 0; mt < 4; ++mt)
#pragma unroll
    for (int j = 0; j < 3; ++j) acc[mt][j] = vzero;

  const bf16_t* xrow[4];
#pragma unroll
  for (int mt = 0; mt < 4; ++mt)
    xrow[mt] = xs + (mrow + 16 * mt) * XSTRIDE + akofs;
  const bf16_t* brow[3];
#pragma unroll
  for (int j = 0; j < 3; ++j)
    brow[j] = w1t + (size_t)(nbase + 16 * j + ncol) * HD + bkofs;

  for (int ks = 0; ks < NKSTEPS; ++ks) {
    const int k0 = ks * KSTEP;
    V16U A[4];
#pragma unroll
    for (int mt = 0; mt < 4; ++mt) {
      A[mt].h[0] = *(const v8bf*)(xrow[mt] + k0);        // K = k0+akofs .. +7
      A[mt].h[1] = *(const v8bf*)(xrow[mt] + k0 + 16);   // K = k0+16+akofs .. +7
    }
#pragma unroll
    for (int j = 0; j < 3; ++j) {
      V16U Bv;                                           // two 16B chunks per lane
      Bv.h[0] = *(const v8bf*)(brow[j] + k0);            // K = k0+bkofs .. +7
      Bv.h[1] = *(const v8bf*)(brow[j] + k0 + 16);       // K = k0+16+bkofs .. +7
#pragma unroll
      for (int mt = 0; mt < 4; ++mt)
        acc[mt][j] = __builtin_amdgcn_wmma_f32_16x16x32_bf16(
            false, A[mt].v, false, Bv.v, (short)0, acc[mt][j], false, false);
    }
  }

  // Epilogue: bias + relu, dot with W2 column, reduce over hidden dim.
  float part[4][8];
#pragma unroll
  for (int mt = 0; mt < 4; ++mt)
#pragma unroll
    for (int r = 0; r < 8; ++r) part[mt][r] = 0.f;
#pragma unroll
  for (int j = 0; j < 3; ++j) {
    const int   n   = nbase + 16 * j + ncol;
    const float bv  = b1[n];
    const float w2v = W2[n];
#pragma unroll
    for (int mt = 0; mt < 4; ++mt)
#pragma unroll
      for (int r = 0; r < 8; ++r) {
        float h = acc[mt][j][r] + bv;
        h = h > 0.f ? h : 0.f;
        part[mt][r] += h * w2v;
      }
  }
  // C/D layout: within an M-tile, m = r + 8*(lane>>4), n = lane&15.
  // Reduce over the 16 lanes of each half (same m) via wave32 shuffles.
#pragma unroll
  for (int mt = 0; mt < 4; ++mt)
#pragma unroll
    for (int r = 0; r < 8; ++r) {
      float v = part[mt][r];
      v += __shfl_xor(v, 1, 32);
      v += __shfl_xor(v, 2, 32);
      v += __shfl_xor(v, 4, 32);
      v += __shfl_xor(v, 8, 32);
      if ((lane & 15) == 0)
        atomicAdd(&red[mt * 16 + (lane >> 4) * 8 + r], v);
    }
  __syncthreads();

  if (tid < MT) {
    float s = red[tid] + b2[0];
    float g = 1.f / (1.f + __expf(-s));
    gate[(size_t)b * PAIRS + p0 + tid] = g;    // [b][t][i], p = t*ILEN + i
  }
}

// ---------------------------------------------------------------------------
// Kernel 3: gated cross-attention. One wave per (b, h, t); wave32 shuffles
// for softmax reductions; gate applied post-softmax.
// ---------------------------------------------------------------------------
__global__ __launch_bounds__(256) void k_attn(
    const float* __restrict__ q, const float* __restrict__ kk,
    const float* __restrict__ vv, const float* __restrict__ gate,
    float* __restrict__ out) {
  __shared__ float qS[8][DHEAD];
  __shared__ float pS[8][ILEN + 3];

  const int gid  = blockIdx.x;                 // 0 .. 8*12*8-1
  const int b    = gid / (NHEAD * 8);
  const int rem  = gid - b * (NHEAD * 8);
  const int h    = rem / 8;
  const int tg   = rem - h * 8;
  const int wave = threadIdx.x >> 5;
  const int lane = threadIdx.x & 31;
  const int t    = tg * 8 + wave;

  const float* qrow = q + ((size_t)(b * TLEN + t)) * HD + h * DHEAD;
  qS[wave][lane]      = qrow[lane];
  qS[wave][lane + 32] = qrow[lane + 32];
  __syncthreads();

  const float* kbase = kk + ((size_t)b * ILEN) * HD + h * DHEAD;
  float sc[7];
  float mymax = -1e30f;
#pragma unroll
  for (int ii = 0; ii < 7; ++ii) {
    const int i = lane + ii * 32;
    float s = -1e30f;
    if (i < ILEN) {
      const float* krow = kbase + (size_t)i * HD;
      float acc = 0.f;
#pragma unroll
      for (int d = 0; d < DHEAD; d += 4) {
        float4 kv = *(const float4*)(krow + d);
        acc += qS[wave][d] * kv.x + qS[wave][d + 1] * kv.y +
               qS[wave][d + 2] * kv.z + qS[wave][d + 3] * kv.w;
      }
      s = acc * 0.125f;                        // 1/sqrt(64)
    }
    sc[ii] = s;
    mymax = fmaxf(mymax, s);
  }
#pragma unroll
  for (int off = 16; off >= 1; off >>= 1)
    mymax = fmaxf(mymax, __shfl_xor(mymax, off, 32));

  float ex[7];
  float mysum = 0.f;
#pragma unroll
  for (int ii = 0; ii < 7; ++ii) {
    const int i = lane + ii * 32;
    ex[ii] = (i < ILEN) ? __expf(sc[ii] - mymax) : 0.f;
    mysum += ex[ii];
  }
#pragma unroll
  for (int off = 16; off >= 1; off >>= 1)
    mysum += __shfl_xor(mysum, off, 32);
  const float inv = 1.f / mysum;

  const float* grow = gate + ((size_t)(b * TLEN + t)) * ILEN;
#pragma unroll
  for (int ii = 0; ii < 7; ++ii) {
    const int i = lane + ii * 32;
    if (i < ILEN) pS[wave][i] = ex[ii] * inv * grow[i];
  }
  __syncthreads();

  const float* vbase = vv + ((size_t)b * ILEN) * HD + h * DHEAD;
  float a0 = 0.f, a1 = 0.f;
  for (int i = 0; i < ILEN; ++i) {
    const float p = pS[wave][i];
    a0 += p * vbase[(size_t)i * HD + lane];
    a1 += p * vbase[(size_t)i * HD + lane + 32];
  }
  float* orow = out + ((size_t)(b * TLEN + t)) * HD + h * DHEAD;
  orow[lane]      = a0;
  orow[lane + 32] = a1;
}

// ---------------------------------------------------------------------------
extern "C" void kernel_launch(void* const* d_in, const int* in_sizes, int n_in,
                              void* d_out, int out_size, void* d_ws, size_t ws_size,
                              hipStream_t stream) {
  (void)in_sizes; (void)n_in; (void)out_size; (void)ws_size;
  const float* text  = (const float*)d_in[0];
  const float* image = (const float*)d_in[1];
  const float* Wq = (const float*)d_in[2];
  const float* bq = (const float*)d_in[3];
  const float* Wk = (const float*)d_in[4];
  const float* bk = (const float*)d_in[5];
  const float* Wv = (const float*)d_in[6];
  const float* bv = (const float*)d_in[7];
  const float* Wt = (const float*)d_in[8];
  const float* bt = (const float*)d_in[9];
  const float* Wi = (const float*)d_in[10];
  const float* bi = (const float*)d_in[11];
  const float* W1 = (const float*)d_in[12];
  const float* b1 = (const float*)d_in[13];
  const float* W2 = (const float*)d_in[14];
  const float* b2 = (const float*)d_in[15];
  float* out = (float*)d_out;

  // Workspace layout (bytes, all 256B aligned)
  const size_t TE_B   = (size_t)BSZ * TLEN * HD * 4;   // 1.50 MB
  const size_t IE_B   = (size_t)BSZ * ILEN * HD * 4;   // 4.62 MB
  const size_t GATE_B = (size_t)BSZ * PAIRS * 4;       // 0.39 MB
  char* ws = (char*)d_ws;
  float*  te   = (float*)(ws);                 ws += TE_B;
  float*  ie   = (float*)(ws);                 ws += IE_B;
  float*  qq   = (float*)(ws);                 ws += TE_B;
  float*  kk   = (float*)(ws);                 ws += IE_B;
  float*  vv   = (float*)(ws);                 ws += IE_B;
  float*  gate = (float*)(ws);                 ws += GATE_B;
  bf16_t* w1t  = (bf16_t*)(ws);

  // Prep: W1 -> bf16 transposed
  k_transpose_w1<<<(HD * HD) / 256, 256, 0, stream>>>(W1, w1t);

  // Projections (rows are multiples of 8: 512 and 1576)
  k_proj<<<(BSZ * TLEN) / 8, 256, 0, stream>>>(text,  Wt, bt, te);
  k_proj<<<(BSZ * ILEN) / 8, 256, 0, stream>>>(image, Wi, bi, ie);
  k_proj<<<(BSZ * TLEN) / 8, 256, 0, stream>>>(text,  Wq, bq, qq);
  k_proj<<<(BSZ * ILEN) / 8, 256, 0, stream>>>(image, Wk, bk, kk);
  k_proj<<<(BSZ * ILEN) / 8, 256, 0, stream>>>(image, Wv, bv, vv);

  // Fused gate MLP (WMMA bf16): the 119 GFLOP hot loop
  k_gate<<<dim3(GATE_TILES, BSZ), 512, GATE_LDS_BYTES, stream>>>(
      te, ie, w1t, b1, W2, b2, gate);

  // Gated cross-attention
  k_attn<<<BSZ * NHEAD * (TLEN / 8), 256, 0, stream>>>(qq, kk, vv, gate, out);
}
